// HyperGNN2D_52226802319464
// MI455X (gfx1250) — matrix operand order repaired and checked by reading, compiled, not run on hardware
//
#include <hip/hip_runtime.h>
#include <math.h>

// CDNA5 / gfx1250 wave32 WMMA types
typedef __attribute__((ext_vector_type(16))) _Float16 v16h;
typedef __attribute__((ext_vector_type(8)))  _Float16 v8h;
typedef __attribute__((ext_vector_type(8)))  float    v8f;

__device__ __forceinline__ float actf(float v, int act) {
  switch (act) {
    case 1:  return v >= 0.f ? v : 0.01f * v;       // leaky 0.01
    case 2:  return tanhf(v);
    case 3:  return 1.f / (1.f + expf(-v));         // sigmoid
    default: return v;
  }
}

// ---------------------------------------------------------------------------
// GEMM on pre-staged f16 operands:
//   A  : [M][Kp] f16 row-major, Kp % 32 == 0, tail K zero-padded
//   Bt : [N][Kp] f16 K-major (transposed weight), zero-padded
// One wave = one 16x16 C tile; inner loop = 2x16B A loads + 2x16B B loads + wmma.
// Row/col are CLAMPED for loads (always in-bounds), stores are guarded -> no
// per-lane exec masking in the hot loop.
// A-frag lane layout (ISA 7.12.2): lane L row L%16; K = k0+(L/16)*8+[0..8) and +16.
// B-frag: lane L col L%16; K = k0+(L/16)*8+[0..8) and +16 (same as A).
// ---------------------------------------------------------------------------
extern "C" __global__ __launch_bounds__(256)
void gemm_f16_wmma(const _Float16* __restrict__ A,
                   const _Float16* __restrict__ Bt,
                   const float* __restrict__ bias,
                   float* __restrict__ C,
                   int M, int N, int Kp, int act) {
  const int wave   = threadIdx.x >> 5;
  const int lane   = threadIdx.x & 31;
  const int laneHi = lane >> 4;
  const int lmod   = lane & 15;
  const int tileM  = blockIdx.x * 8 + wave;
  const int tileN  = blockIdx.y;
  if (tileM * 16 >= M) return;

  int row = tileM * 16 + lmod; if (row >= M) row = M - 1;      // clamp
  int col = tileN * 16 + lmod;
  int colc = col < N ? col : N - 1;                            // clamp

  const _Float16* Ar = A  + (size_t)row  * Kp;
  const _Float16* Br = Bt + (size_t)colc * Kp;

  v8f acc = {};
  for (int k0 = 0; k0 < Kp; k0 += 32) {
    __builtin_prefetch(Ar + k0 + 128, 0, 3);                   // global_prefetch_b8
    v8h lo = *(const v8h*)(Ar + k0 + laneHi * 8);
    v8h hi = *(const v8h*)(Ar + k0 + 16 + laneHi * 8);
    v16h a = __builtin_shufflevector(lo, hi, 0,1,2,3,4,5,6,7,8,9,10,11,12,13,14,15);
    v8h blo = *(const v8h*)(Br + k0 + laneHi * 8);
    v8h bhi = *(const v8h*)(Br + k0 + 16 + laneHi * 8);
    v16h b = __builtin_shufflevector(blo, bhi, 0,1,2,3,4,5,6,7,8,9,10,11,12,13,14,15);
    acc = __builtin_amdgcn_wmma_f32_16x16x32_f16(false, a, false, b, (short)0, acc,
                                                 false, false);
  }
#pragma unroll
  for (int r = 0; r < 8; ++r) {
    int m = tileM * 16 + laneHi * 8 + r;
    int n = tileN * 16 + lmod;
    if (m < M && n < N) {
      float v = acc[r];
      if (bias) v += bias[n];
      C[(size_t)m * N + n] = actf(v, act);
    }
  }
}

// ---------------------------------------------------------------------------
// NNConv edge-function as fused GEMM, fully register-resident A:
//   msg[e,o] = sum_{p,i} pin16[e,p]*xg[e,i]*Bt[o, p*32+i] + sum_i xg[e,i]*Bt[o,512+i]
// K = 544 = 17 steps; A fragment built per step as (scalar pin_p) * xg slices.
// pin16: [E][16] f16;  xg: [E][32] f16 (pre-gathered net[dst(e)]);  Bt: [32][544] f16.
// ---------------------------------------------------------------------------
extern "C" __global__ __launch_bounds__(256)
void nnconv_wmma(const _Float16* __restrict__ pin16,
                 const _Float16* __restrict__ xg,
                 const _Float16* __restrict__ Bt,
                 float* __restrict__ msg, int E) {
  const int wave   = threadIdx.x >> 5;
  const int lane   = threadIdx.x & 31;
  const int laneHi = lane >> 4;
  const int lmod   = lane & 15;
  const int tileM  = blockIdx.x * 8 + wave;
  if (tileM * 16 >= E) return;

  int row = tileM * 16 + lmod; if (row >= E) row = E - 1;      // clamp
  const int col = blockIdx.y * 16 + lmod;                      // N = 32, always valid

  // Preload per-lane operand slices (all subsequent math is in registers).
  v8h  xlo = *(const v8h*)(xg + (size_t)row * 32 + laneHi * 8);        // i: laneHi*8..+8
  v8h  xhi = *(const v8h*)(xg + (size_t)row * 32 + 16 + laneHi * 8);   // i: 16+laneHi*8..+8
  v16h pv  = *(const v16h*)(pin16 + (size_t)row * 16);                 // 16 pin feats

  const _Float16* Br = Bt + (size_t)col * 544;

  v8f acc = {};
#pragma unroll
  for (int s = 0; s < 17; ++s) {
    const int k0 = s * 32;
    v8h alo, ahi;
    if (s < 16) { _Float16 p = pv[s]; alo = xlo * p; ahi = xhi * p; }
    else        { alo = xlo; ahi = xhi; }                      // bias rows: A = xg
    v16h a = __builtin_shufflevector(alo, ahi, 0,1,2,3,4,5,6,7,8,9,10,11,12,13,14,15);
    v8h blo = *(const v8h*)(Br + k0 + laneHi * 8);
    v8h bhi = *(const v8h*)(Br + k0 + 16 + laneHi * 8);
    v16h b = __builtin_shufflevector(blo, bhi, 0,1,2,3,4,5,6,7,8,9,10,11,12,13,14,15);
    acc = __builtin_amdgcn_wmma_f32_16x16x32_f16(false, a, false, b, (short)0, acc,
                                                 false, false);
  }
#pragma unroll
  for (int r = 0; r < 8; ++r) {
    int m = tileM * 16 + laneHi * 8 + r;
    int n = blockIdx.y * 16 + lmod;
    if (m < E) msg[(size_t)m * 32 + n] = acc[r];
  }
}

// --------------------------- staging / conversion --------------------------
// f32 [M][K1]+[M][K2] (concat, optional rowscale) -> f16 [M][Kp], zero-padded.
extern "C" __global__ void k_a16(const float* __restrict__ A1, int K1,
                                 const float* __restrict__ A2, int K2,
                                 const float* __restrict__ rowscale,
                                 _Float16* __restrict__ dst, int M, int Kp) {
  int i = blockIdx.x * blockDim.x + threadIdx.x;
  if (i < M * Kp) {
    int m = i / Kp, k = i % Kp;
    float v = 0.f;
    if (k < K1)           v = A1[(size_t)m * K1 + k];
    else if (k < K1 + K2) v = A2[(size_t)m * K2 + (k - K1)];
    if (rowscale) v *= rowscale[m];
    dst[i] = (_Float16)v;
  }
}

// f32 weight [K][N] -> f16 K-major [N][Kp], zero-padded.
extern "C" __global__ void k_bt16(const float* __restrict__ B, _Float16* __restrict__ Bt,
                                  int K, int N, int Kp) {
  int i = blockIdx.x * blockDim.x + threadIdx.x;
  if (i < N * Kp) {
    int n = i / Kp, k = i % Kp;
    Bt[i] = (_Float16)(k < K ? B[(size_t)k * N + n] : 0.f);
  }
}

// NNConv weight: lin2_w[16][1024] + lin2_b[1024] -> Bt [32][544] f16 K-major.
extern "C" __global__ void k_nnbt(const float* __restrict__ l2w,
                                  const float* __restrict__ l2b,
                                  _Float16* __restrict__ Bt) {
  int i = blockIdx.x * blockDim.x + threadIdx.x;
  if (i < 32 * 544) {
    int o = i / 544, k = i % 544;
    float v = (k < 512) ? l2w[(size_t)(k >> 5) * 1024 + (size_t)(k & 31) * 32 + o]
                        : l2b[(size_t)(k - 512) * 32 + o];
    Bt[i] = (_Float16)v;
  }
}

// xg[e][c] = f16(net[dst(e)][c])
extern "C" __global__ void k_gather16(const int* __restrict__ dn,
                                      const float* __restrict__ net,
                                      _Float16* __restrict__ xg, int E32) {
  int i = blockIdx.x * blockDim.x + threadIdx.x;
  if (i < E32) {
    int e = i >> 5, c = i & 31;
    xg[i] = (_Float16)net[(size_t)dn[e] * 32 + c];
  }
}

// ------------------------------ small kernels ------------------------------
extern "C" __global__ void k_fill(float* p, float v, int n) {
  int i = blockIdx.x * blockDim.x + threadIdx.x;
  if (i < n) p[i] = v;
}

extern "C" __global__ void k_count(const int* __restrict__ idx, float* deg, int E) {
  int i = blockIdx.x * blockDim.x + threadIdx.x;
  if (i < E) atomicAdd(&deg[idx[i]], 1.f);
}

extern "C" __global__ void k_deg_fin(float* deg, float* isq, int n) {
  int i = blockIdx.x * blockDim.x + threadIdx.x;
  if (i < n) { float d = fmaxf(deg[i], 1.f); deg[i] = d; isq[i] = rsqrtf(d); }
}

extern "C" __global__ void k_attn(const float* __restrict__ Hf,
                                  const float* __restrict__ aL,
                                  const float* __restrict__ aR,
                                  float* el, float* er, int n4) {
  int i = blockIdx.x * blockDim.x + threadIdx.x;
  if (i < n4) {
    int node = i >> 2, h = i & 3;
    float sl = 0.f, sr = 0.f;
#pragma unroll
    for (int f = 0; f < 8; ++f) {
      float hv = Hf[(size_t)node * 32 + h * 8 + f];
      sl += hv * aL[h * 8 + f];
      sr += hv * aR[h * 8 + f];
    }
    el[i] = sl; er[i] = sr;
  }
}

__device__ __forceinline__ void atomicMaxF(float* addr, float val) {
  unsigned int* ua = (unsigned int*)addr;
  unsigned int old = __float_as_uint(*addr);
  while (__uint_as_float(old) < val) {
    unsigned int assumed = old;
    old = atomicCAS(ua, assumed, __float_as_uint(val));
    if (old == assumed) break;
  }
}

extern "C" __global__ void k_gat_max(const int* __restrict__ src, const int* __restrict__ dst,
                                     const float* __restrict__ el, const float* __restrict__ er,
                                     float* redm, int E4) {
  int i = blockIdx.x * blockDim.x + threadIdx.x;
  if (i < E4) {
    int e = i >> 2, h = i & 3;
    int s = src[e], d = dst[e];
    float sc = el[s * 4 + h] + er[d * 4 + h];
    sc = sc >= 0.f ? sc : 0.2f * sc;
    atomicMaxF(&redm[d * 4 + h], sc);
  }
}

extern "C" __global__ void k_gat_sum(const int* __restrict__ src, const int* __restrict__ dst,
                                     const float* __restrict__ el, const float* __restrict__ er,
                                     const float* __restrict__ redm, float* reds, int E4) {
  int i = blockIdx.x * blockDim.x + threadIdx.x;
  if (i < E4) {
    int e = i >> 2, h = i & 3;
    int s = src[e], d = dst[e];
    float sc = el[s * 4 + h] + er[d * 4 + h];
    sc = sc >= 0.f ? sc : 0.2f * sc;
    atomicAdd(&reds[d * 4 + h], expf(sc - redm[d * 4 + h]));
  }
}

extern "C" __global__ void k_gat_scatter(const int* __restrict__ src, const int* __restrict__ dst,
                                         const float* __restrict__ el, const float* __restrict__ er,
                                         const float* __restrict__ redm, const float* __restrict__ reds,
                                         const float* __restrict__ Hf, float* nodenew,
                                         int chan, int E32) {
  int i = blockIdx.x * blockDim.x + threadIdx.x;
  if (i < E32) {
    int e = i >> 5, c = i & 31, h = c >> 3, f = c & 7;
    int s = src[e], d = dst[e];
    float sc = el[s * 4 + h] + er[d * 4 + h];
    sc = sc >= 0.f ? sc : 0.2f * sc;
    float a = expf(sc - redm[d * 4 + h]) / fmaxf(reds[d * 4 + h], 1e-9f);
    atomicAdd(&nodenew[(size_t)d * 96 + h * 16 + chan * 8 + f],
              Hf[(size_t)s * 32 + c] * a);
  }
}

extern "C" __global__ void k_scatter32(const int* __restrict__ src, const int* __restrict__ dst,
                                       const float* __restrict__ x, float* acc, int E32) {
  int i = blockIdx.x * blockDim.x + threadIdx.x;
  if (i < E32) {
    int e = i >> 5, c = i & 31;
    atomicAdd(&acc[(size_t)dst[e] * 32 + c], x[(size_t)src[e] * 32 + c]);
  }
}

extern "C" __global__ void k_net_fin(const float* __restrict__ acc, const float* __restrict__ isq,
                                     const float* __restrict__ b, float* out, int n32) {
  int i = blockIdx.x * blockDim.x + threadIdx.x;
  if (i < n32) { int n = i >> 5, c = i & 31; out[i] = tanhf(acc[i] * isq[n] + b[c]); }
}

extern "C" __global__ void k_msg_scatter(const int* __restrict__ ndst,
                                         const float* __restrict__ msg,
                                         float* nodenew, int E32) {
  int i = blockIdx.x * blockDim.x + threadIdx.x;
  if (i < E32) {
    int e = i >> 5, c = i & 31;
    atomicAdd(&nodenew[(size_t)ndst[e] * 96 + 64 + c], msg[i]);
  }
}

extern "C" __global__ void k_node_fin(float* nb_, const float* __restrict__ gbias,
                                      const float* __restrict__ nbias,
                                      const float* __restrict__ deg, int n96) {
  int i = blockIdx.x * blockDim.x + threadIdx.x;
  if (i < n96) {
    int n = i / 96, c = i % 96;
    float v = nb_[i];
    if (c < 64) v += gbias[((c >> 4) << 3) + (c & 7)];
    else        v = v / deg[n] + nbias[c - 64];
    nb_[i] = tanhf(v);
  }
}

// ---------------------------------------------------------------------------
extern "C" void kernel_launch(void* const* d_in, const int* in_sizes, int n_in,
                              void* d_out, int out_size, void* d_ws, size_t ws_size,
                              hipStream_t stream) {
  const float* in_node    = (const float*)d_in[0];
  const float* in_net     = (const float*)d_in[1];
  const float* in_pin     = (const float*)d_in[2];
  const float* node_lin_w = (const float*)d_in[3];  const float* node_lin_b = (const float*)d_in[4];
  const float* net_lin_w  = (const float*)d_in[5];  const float* net_lin_b  = (const float*)d_in[6];
  const float* pin_lin_w  = (const float*)d_in[7];  const float* pin_lin_b  = (const float*)d_in[8];
  const float* gat_fc_w   = (const float*)d_in[9];
  const float* gat_attn_l = (const float*)d_in[10]; const float* gat_attn_r = (const float*)d_in[11];
  const float* gat_bias   = (const float*)d_in[12];
  const float* gconv_w    = (const float*)d_in[13]; const float* gconv_b    = (const float*)d_in[14];
  const float* lin2_w     = (const float*)d_in[15]; const float* lin2_b     = (const float*)d_in[16];
  const float* nnconv_bp  = (const float*)d_in[17];
  const float* out1_w = (const float*)d_in[18]; const float* out1_b = (const float*)d_in[19];
  const float* out2_w = (const float*)d_in[20]; const float* out2_b = (const float*)d_in[21];
  const float* out3_w = (const float*)d_in[22]; const float* out3_b = (const float*)d_in[23];
  const int* pins_src = (const int*)d_in[24];
  const int* pins_dst = (const int*)d_in[25];
  const int* grid_src = (const int*)d_in[26];
  const int* grid_dst = (const int*)d_in[27];

  const int Nn = in_sizes[0] / 16, Nnet = in_sizes[1] / 8;
  const int Ep = in_sizes[24], Eg = in_sizes[26] / 2;

  size_t off = 0;
  auto alloc = [&](size_t bytes) -> void* {
    void* p = (char*)d_ws + off;
    off += bytes; off = (off + 255) & ~(size_t)255;
    return p;
  };
  // f32 state
  float* node_a  = (float*)alloc((size_t)Nn * 96 * 4);
  float* node_b  = (float*)alloc((size_t)Nn * 96 * 4);
  float* net_a   = (float*)alloc((size_t)Nnet * 32 * 4);
  float* net_b   = (float*)alloc((size_t)Nnet * 32 * 4);
  float* pinH    = (float*)alloc((size_t)Ep * 16 * 4);
  float* Hfeat   = (float*)alloc((size_t)Nn * 32 * 4);
  float* el      = (float*)alloc((size_t)Nn * 4 * 4);
  float* er      = (float*)alloc((size_t)Nn * 4 * 4);
  float* redm    = (float*)alloc((size_t)Nn * 4 * 4);
  float* reds    = (float*)alloc((size_t)Nn * 4 * 4);
  float* deg_node = (float*)alloc((size_t)Nn * 4);
  float* isq_node = (float*)alloc((size_t)Nn * 4);
  float* deg_net  = (float*)alloc((size_t)Nnet * 4);
  float* isq_net  = (float*)alloc((size_t)Nnet * 4);
  float* tmp32   = (float*)alloc((size_t)Nn * 32 * 4);
  float* netacc  = (float*)alloc((size_t)Nnet * 32 * 4);
  float* msg     = (float*)alloc((size_t)Ep * 32 * 4);
  float* h1      = (float*)alloc((size_t)Nn * 96 * 4);
  float* h2      = (float*)alloc((size_t)Nn * 96 * 4);
  // f16 staged operands
  _Float16* x_node_in = (_Float16*)alloc((size_t)Nn * 32 * 2);
  _Float16* x_net_in  = (_Float16*)alloc((size_t)Nnet * 32 * 2);
  _Float16* x_pin_in  = (_Float16*)alloc((size_t)Ep * 32 * 2);
  _Float16* node16    = (_Float16*)alloc((size_t)Nn * 96 * 2);
  _Float16* node16s   = (_Float16*)alloc((size_t)Nn * 96 * 2);
  _Float16* cat16     = (_Float16*)alloc((size_t)Nn * 128 * 2);
  _Float16* h16       = (_Float16*)alloc((size_t)Nn * 96 * 2);
  _Float16* pin16     = (_Float16*)alloc((size_t)Ep * 16 * 2);
  _Float16* xg        = (_Float16*)alloc((size_t)Ep * 32 * 2);
  // f16 K-major weights
  _Float16* w_node = (_Float16*)alloc(96 * 32 * 2);
  _Float16* w_net  = (_Float16*)alloc(32 * 32 * 2);
  _Float16* w_pin  = (_Float16*)alloc(16 * 32 * 2);
  _Float16* w_fc   = (_Float16*)alloc(2 * 32 * 96 * 2);
  _Float16* w_gc   = (_Float16*)alloc(2 * 32 * 96 * 2);
  _Float16* w_nn   = (_Float16*)alloc(2 * 32 * 544 * 2);
  _Float16* w_o1   = (_Float16*)alloc(96 * 128 * 2);
  _Float16* w_o2   = (_Float16*)alloc(96 * 96 * 2);
  _Float16* w_o3   = (_Float16*)alloc(4 * 96 * 2);

  auto fill = [&](float* p, float v, size_t n) {
    k_fill<<<(int)((n + 255) / 256), 256, 0, stream>>>(p, v, (int)n);
  };
  auto bt = [&](const float* B, _Float16* Bt, int K, int N, int Kp) {
    int n = N * Kp;
    k_bt16<<<(n + 255) / 256, 256, 0, stream>>>(B, Bt, K, N, Kp);
  };
  auto a16 = [&](const float* A1, int K1, const float* A2, int K2, const float* rsc,
                 _Float16* dst, int M, int Kp) {
    int n = M * Kp;
    k_a16<<<(n + 255) / 256, 256, 0, stream>>>(A1, K1, A2, K2, rsc, dst, M, Kp);
  };
  auto gemm = [&](const _Float16* A, const _Float16* Bt, const float* bias,
                  float* Cp, int M, int N, int Kp, int act) {
    dim3 g((M + 127) / 128, (N + 15) / 16);
    gemm_f16_wmma<<<g, 256, 0, stream>>>(A, Bt, bias, Cp, M, N, Kp, act);
  };

  // ---- weight staging (once) ----
  bt(node_lin_w, w_node, 16, 96, 32);
  bt(net_lin_w,  w_net,   8, 32, 32);
  bt(pin_lin_w,  w_pin,   8, 16, 32);
  for (int i = 0; i < 2; ++i) {
    bt(gat_fc_w + (size_t)i * 96 * 32, w_fc + (size_t)i * 32 * 96, 96, 32, 96);
    bt(gconv_w  + (size_t)i * 96 * 32, w_gc + (size_t)i * 32 * 96, 96, 32, 96);
    k_nnbt<<<(32 * 544 + 255) / 256, 256, 0, stream>>>(
        lin2_w + (size_t)i * 16 * 1024, lin2_b + (size_t)i * 1024,
        w_nn + (size_t)i * 32 * 544);
  }
  bt(out1_w, w_o1, 112, 96, 128);
  bt(out2_w, w_o2,  96, 96,  96);
  bt(out3_w, w_o3,  96,  4,  96);

  // ---- input staging ----
  a16(in_node, 16, nullptr, 0, nullptr, x_node_in, Nn,  32);
  a16(in_net,   8, nullptr, 0, nullptr, x_net_in, Nnet, 32);
  a16(in_pin,   8, nullptr, 0, nullptr, x_pin_in, Ep,   32);

  // ---- degrees over pin graph ----
  fill(deg_node, 0.f, Nn); fill(deg_net, 0.f, Nnet);
  k_count<<<(Ep + 255) / 256, 256, 0, stream>>>(pins_src, deg_node, Ep);
  k_count<<<(Ep + 255) / 256, 256, 0, stream>>>(pins_dst, deg_net, Ep);
  k_deg_fin<<<(Nn + 255) / 256, 256, 0, stream>>>(deg_node, isq_node, Nn);
  k_deg_fin<<<(Nnet + 255) / 256, 256, 0, stream>>>(deg_net, isq_net, Nnet);

  // ---- input projections (leaky 0.01) ----
  gemm(x_node_in, w_node, node_lin_b, node_a, Nn,  96, 32, 1);
  gemm(x_net_in,  w_net,  net_lin_b,  net_a, Nnet, 32, 32, 1);
  gemm(x_pin_in,  w_pin,  pin_lin_b,  pinH,  Ep,   16, 32, 1);
  a16(pinH, 16, nullptr, 0, nullptr, pin16, Ep, 16);   // indexed access only, Kp=16 ok

  float* node_cur = node_a; float* node_new = node_b;
  float* net_cur  = net_a;  float* net_new  = net_b;

  for (int i = 0; i < 2; ++i) {
    const _Float16* wfc = w_fc + (size_t)i * 32 * 96;
    const _Float16* wgc = w_gc + (size_t)i * 32 * 96;
    const _Float16* wnn = w_nn + (size_t)i * 32 * 544;
    const float* aL    = gat_attn_l + (size_t)i * 32;
    const float* aR    = gat_attn_r + (size_t)i * 32;
    const float* gbias = gat_bias   + (size_t)i * 32;
    const float* gb    = gconv_b    + (size_t)i * 32;
    const float* nbb   = nnconv_bp  + (size_t)i * 32;

    a16(node_cur, 96, nullptr, 0, nullptr, node16, Nn, 96);
    fill(node_new, 0.f, (size_t)Nn * 96);

    // GAT: shared fc projection + attention scores
    gemm(node16, wfc, nullptr, Hfeat, Nn, 32, 96, 0);
    k_attn<<<(Nn * 4 + 255) / 256, 256, 0, stream>>>(Hfeat, aL, aR, el, er, Nn * 4);

    for (int j = 0; j < 2; ++j) {
      const int* gs = grid_src + (size_t)j * Eg;
      const int* gd = grid_dst + (size_t)j * Eg;
      fill(redm, -1e30f, (size_t)Nn * 4);
      fill(reds, 0.f, (size_t)Nn * 4);
      k_gat_max<<<(Eg * 4 + 255) / 256, 256, 0, stream>>>(gs, gd, el, er, redm, Eg * 4);
      k_gat_sum<<<(Eg * 4 + 255) / 256, 256, 0, stream>>>(gs, gd, el, er, redm, reds, Eg * 4);
      int e32 = Eg * 32;
      k_gat_scatter<<<(e32 + 255) / 256, 256, 0, stream>>>(gs, gd, el, er, redm, reds,
                                                           Hfeat, node_new, j, e32);
    }

    // GraphConv node->net with D^-1/2 pre-scale folded into f16 staging
    a16(node_cur, 96, nullptr, 0, isq_node, node16s, Nn, 96);
    gemm(node16s, wgc, nullptr, tmp32, Nn, 32, 96, 0);
    fill(netacc, 0.f, (size_t)Nnet * 32);
    k_scatter32<<<(Ep * 32 + 255) / 256, 256, 0, stream>>>(pins_src, pins_dst, tmp32,
                                                           netacc, Ep * 32);
    k_net_fin<<<(Nnet * 32 + 255) / 256, 256, 0, stream>>>(netacc, isq_net, gb,
                                                           net_new, Nnet * 32);

    // NNConv net->node: pre-gather, fused WMMA edge GEMM, scatter-mean
    k_gather16<<<(Ep * 32 + 255) / 256, 256, 0, stream>>>(pins_dst, net_cur, xg, Ep * 32);
    {
      dim3 g((Ep + 127) / 128, 2);
      nnconv_wmma<<<g, 256, 0, stream>>>(pin16, xg, wnn, msg, Ep);
    }
    k_msg_scatter<<<(Ep * 32 + 255) / 256, 256, 0, stream>>>(pins_src, msg, node_new,
                                                             Ep * 32);

    k_node_fin<<<(Nn * 96 + 255) / 256, 256, 0, stream>>>(node_new, gbias, nbb,
                                                          deg_node, Nn * 96);

    { float* t = node_cur; node_cur = node_new; node_new = t; }
    { float* t = net_cur;  net_cur  = net_new;  net_new  = t; }
  }

  // ---- output MLP ----
  a16(in_node, 16, node_cur, 96, nullptr, cat16, Nn, 128);   // concat, Kp=128
  gemm(cat16, w_o1, out1_b, h1, Nn, 96, 128, 2);
  a16(h1, 96, nullptr, 0, nullptr, h16, Nn, 96);
  gemm(h16, w_o2, out2_b, h2, Nn, 96, 96, 2);
  a16(h2, 96, nullptr, 0, nullptr, h16, Nn, 96);
  gemm(h16, w_o3, out3_b, (float*)d_out, Nn, 4, 96, 3);

  (void)ws_size; (void)n_in; (void)out_size;
}
